// MambaBackbone3D_87187836108820
// MI455X (gfx1250) — matrix operand order repaired and checked by reading, not compiled, not run on hardware
//
#include <hip/hip_runtime.h>

// ---------------- CDNA5 WMMA types ----------------
typedef __attribute__((ext_vector_type(16))) __bf16 v16bf;
typedef __attribute__((ext_vector_type(8)))  float  v8f;

#define EPS 1e-3f

// =====================================================================
// Scatter active voxels into dense grid (deterministic: max index wins)
// =====================================================================
__global__ void scatter_max_kernel(const int* __restrict__ coords, int n,
                                   int H, int W, int* __restrict__ winner) {
  int i = blockIdx.x * blockDim.x + threadIdx.x;
  if (i >= n) return;
  int z = coords[i * 4 + 1], y = coords[i * 4 + 2], x = coords[i * 4 + 3];
  atomicMax(&winner[(z * H + y) * W + x], i);
}

// writes 4 real channels into 16-channel zero-padded dense input
__global__ void scatter_write_kernel(const float* __restrict__ vf,
                                     const int* __restrict__ coords, int n,
                                     int H, int W,
                                     const int* __restrict__ winner,
                                     float* __restrict__ dst /*[vox][16]*/) {
  int i = blockIdx.x * blockDim.x + threadIdx.x;
  if (i >= n) return;
  int z = coords[i * 4 + 1], y = coords[i * 4 + 2], x = coords[i * 4 + 3];
  int vox = (z * H + y) * W + x;
  if (winner[vox] == i) {
    dst[vox * 16 + 0] = vf[i * 4 + 0];
    dst[vox * 16 + 1] = vf[i * 4 + 1];
    dst[vox * 16 + 2] = vf[i * 4 + 2];
    dst[vox * 16 + 3] = vf[i * 4 + 3];
  }
}

__global__ void compact_kernel(const int* __restrict__ winner, int n,
                               unsigned char* __restrict__ mask,
                               int* __restrict__ list, int* __restrict__ count) {
  int i = blockIdx.x * blockDim.x + threadIdx.x;
  if (i >= n) return;
  int a = winner[i] >= 0;
  mask[i] = (unsigned char)a;
  if (a) { int p = atomicAdd(count, 1); list[p] = i; }
}

// =====================================================================
// Mask max-pool (output active set for strided sparse convs) + compaction
// =====================================================================
__global__ void maskpool_kernel(const unsigned char* __restrict__ srcM,
                                int Di, int Hi, int Wi,
                                unsigned char* __restrict__ dstM,
                                int Do, int Ho, int Wo,
                                int* __restrict__ list, int* __restrict__ count,
                                int kz, int ky, int kx, int sz, int sy, int sx,
                                int pz, int py, int px) {
  int vox = blockIdx.x * blockDim.x + threadIdx.x;
  int n = Do * Ho * Wo;
  if (vox >= n) return;
  int x = vox % Wo; int t = vox / Wo; int y = t % Ho; int z = t / Ho;
  int act = 0;
  for (int dz = 0; dz < kz; ++dz) {
    int iz = z * sz - pz + dz; if (iz < 0 || iz >= Di) continue;
    for (int dy = 0; dy < ky; ++dy) {
      int iy = y * sy - py + dy; if (iy < 0 || iy >= Hi) continue;
      for (int dx = 0; dx < kx; ++dx) {
        int ix = x * sx - px + dx; if (ix < 0 || ix >= Wi) continue;
        act |= srcM[(iz * Hi + iy) * Wi + ix];
      }
    }
  }
  dstM[vox] = (unsigned char)(act ? 1 : 0);
  if (act) { int p = atomicAdd(count, 1); list[p] = vox; }
}

// =====================================================================
// Pack conv weights [Cout][CinReal][taps] (f32, OIDHW tap-major flattened)
// into the exact per-(kstep, ntile, lane) bf16 B-fragment layout:
//   out[((ks*NT + t4)*32 + lane)*16 + j]  with  k = ks*32 + (lane>>4)*8 + (j&7) + (j>>3)*16
// Zero-pads K beyond Ktot and channels beyond CinReal.
// =====================================================================
__global__ void pack_weights_kernel(const float* __restrict__ wgt, int CinReal,
                                    int lc, int taps, int Cout, int ksteps,
                                    __bf16* __restrict__ out) {
  int tid = blockIdx.x * blockDim.x + threadIdx.x;
  int nt16 = Cout >> 4;
  int total = ksteps * nt16 * 32 * 16;
  if (tid >= total) return;
  int j = tid & 15;
  int lane = (tid >> 4) & 31;
  int rest = tid >> 9;
  int t4 = rest % nt16;
  int ks = rest / nt16;
  int half = lane >> 4, l15 = lane & 15;
  int k = ks * 32 + half * 8 + (j & 7) + ((j >> 3) << 4);
  int nch = t4 * 16 + l15;
  int Ktot = taps << lc;
  float f = 0.f;
  if (k < Ktot) {
    int c = k & ((1 << lc) - 1);
    int t = k >> lc;
    if (c < CinReal) f = wgt[(nch * CinReal + c) * taps + t];
  }
  out[tid] = (__bf16)f;
}

// =====================================================================
// Implicit-GEMM sparse conv, templated so tap decode is constant math and
// the tap index per K-chunk is wave-uniform (K is tap-major, Cin=16/32/64).
// One wave: 16 active voxels x (NT*16) channels, K chunks of 32 (bf16 WMMA).
// =====================================================================
struct F8 { float4 a, b; };

template <int LC, int KZ, int KY, int KX, int NT>
__global__ void __launch_bounds__(128)
conv_wmma(const float* __restrict__ src, int Din, int Hin, int Win,
          const __bf16* __restrict__ bpack, const float* __restrict__ bias,
          float* __restrict__ dst, int Hout, int Wout,
          const int* __restrict__ list, const int* __restrict__ count,
          int dil, int sz, int sy, int sx, int pz, int py, int px) {
  constexpr int TAPS = KZ * KY * KX;
  constexpr int KTOT = TAPS << LC;
  constexpr int KSTEPS = (KTOT + 31) / 32;
  constexpr int COUT = NT * 16;

  const int lane = threadIdx.x & 31;
  const int wave = threadIdx.x >> 5;
  const int tile = blockIdx.x * 4 + wave;
  const int cnt = *count;
  if (tile * 16 >= cnt) return;  // wave-uniform exit
  const int half = lane >> 4, l15 = lane & 15;

  int row = tile * 16 + l15;
  int myvox = (row < cnt) ? list[row] : -1;
  bool rowok = myvox >= 0;
  int bz = 0, by = 0, bx = 0;
  if (rowok) {
    int ox = myvox % Wout; int r = myvox / Wout;
    int oy = r % Hout;     int oz = r / Hout;
    bz = oz * sz - pz; by = oy * sy - py; bx = ox * sx - px;
  }

  // 8 contiguous channels [cb..cb+7] of one tap for this lane's voxel row
  auto loadgrp = [&](int tap, int cb) -> F8 {
    int dz = tap / (KY * KX);
    int rem = tap - dz * (KY * KX);
    int dy = rem / KX;
    int dx = rem - dy * KX;
    int iz = bz + dz, iy = by + dy, ix = bx + dx;
    bool ok = rowok && (tap < TAPS);
    if (dil) {  // lhs_dilation = 2 (inverse conv): odd coords hit holes
      ok = ok && (((iz | iy | ix) & 1) == 0);
      iz >>= 1; iy >>= 1; ix >>= 1;
    }
    ok = ok && iz >= 0 && iz < Din && iy >= 0 && iy < Hin && ix >= 0 && ix < Win;
    F8 r;
    if (ok) {
      const float* p = src + (((size_t)(iz * Hin + iy) * Win + ix) << LC) + cb;
      r.a = *(const float4*)p;
      r.b = *(const float4*)(p + 4);
    } else {
      r.a = float4{0.f, 0.f, 0.f, 0.f};
      r.b = float4{0.f, 0.f, 0.f, 0.f};
    }
    return r;
  };

  v8f acc[NT] = {};

  for (int ks = 0; ks < KSTEPS; ++ks) {
    F8 g0, g1;
    if constexpr (LC == 4) {          // 2 taps per chunk
      g0 = loadgrp(2 * ks, half * 8);
      g1 = loadgrp(2 * ks + 1, half * 8);
    } else if constexpr (LC == 5) {   // 1 tap per chunk
      g0 = loadgrp(ks, half * 8);
      g1 = loadgrp(ks, 16 + half * 8);
    } else {                          // LC==6: half tap per chunk
      int tap = ks >> 1, cb = (ks & 1) << 5;
      g0 = loadgrp(tap, cb + half * 8);
      g1 = loadgrp(tap, cb + 16 + half * 8);
    }
    v16bf A;
    A[0] = (__bf16)g0.a.x; A[1] = (__bf16)g0.a.y;
    A[2] = (__bf16)g0.a.z; A[3] = (__bf16)g0.a.w;
    A[4] = (__bf16)g0.b.x; A[5] = (__bf16)g0.b.y;
    A[6] = (__bf16)g0.b.z; A[7] = (__bf16)g0.b.w;
    A[8] = (__bf16)g1.a.x; A[9] = (__bf16)g1.a.y;
    A[10] = (__bf16)g1.a.z; A[11] = (__bf16)g1.a.w;
    A[12] = (__bf16)g1.b.x; A[13] = (__bf16)g1.b.y;
    A[14] = (__bf16)g1.b.z; A[15] = (__bf16)g1.b.w;

    const __bf16* bp = bpack + ((size_t)(ks * NT) * 32 + lane) * 16;
#pragma unroll
    for (int t4 = 0; t4 < NT; ++t4) {
      v16bf B = *(const v16bf*)(bp + (size_t)t4 * 32 * 16);
      acc[t4] = __builtin_amdgcn_wmma_f32_16x16x32_bf16(
          false, A, false, B, (short)0, acc[t4], false, false);
    }
  }

  // scatter D (+bias): VGPR v, lane-half -> M = v + half*8, N = l15
#pragma unroll
  for (int t4 = 0; t4 < NT; ++t4) {
    int ch = t4 * 16 + l15;
    float bb = bias ? bias[ch] : 0.f;
#pragma unroll
    for (int vv = 0; vv < 8; ++vv) {
      int r = tile * 16 + vv + half * 8;
      if (r < cnt) {
        int vox = list[r];
        dst[(size_t)vox * COUT + ch] = acc[t4][vv] + bb;
      }
    }
  }
}

// =====================================================================
// Generic per-active-site epilogue: t = bn(x + before) + after; relu?
// =====================================================================
__global__ void bn_epilogue(float* __restrict__ x,
                            const float* __restrict__ before,
                            const float* __restrict__ after,
                            const float* __restrict__ g,
                            const float* __restrict__ b,
                            const float* __restrict__ m,
                            const float* __restrict__ v,
                            const int* __restrict__ list,
                            const int* __restrict__ count, int lc, int relu) {
  int tid = blockIdx.x * blockDim.x + threadIdx.x;
  int i = tid >> lc;
  if (i >= *count) return;
  int C = 1 << lc;
  int c = tid & (C - 1);
  int ofs = list[i] * C + c;
  float t = x[ofs];
  if (before) t += before[ofs];
  float s = g[c] * rsqrtf(v[c] + EPS);
  t = (t - m[c]) * s + b[c];
  if (after) t += after[ofs];
  if (relu) t = fmaxf(t, 0.f);
  x[ofs] = t;
}

// Final: [vox=(h*16+w)][64] -> NCHW flat [c*256 + vox]
__global__ void write_out_kernel(const float* __restrict__ src,
                                 float* __restrict__ out) {
  int tid = blockIdx.x * blockDim.x + threadIdx.x;
  if (tid >= 64 * 256) return;
  int c = tid >> 8, vox = tid & 255;
  out[tid] = src[vox * 64 + c];
}

// =====================================================================
// Host sequencing
// =====================================================================
struct Dim { int D, H, W, V; };
struct BNP { const float *g, *b, *m, *v; };

extern "C" void kernel_launch(void* const* d_in, const int* in_sizes, int n_in,
                              void* d_out, int out_size, void* d_ws, size_t ws_size,
                              hipStream_t stream) {
  (void)n_in; (void)out_size; (void)ws_size;
  const float* vf   = (const float*)d_in[0];
  const int* coords = (const int*)d_in[1];
  const int nvox    = in_sizes[0] / 4;

  // param leaves: jax pytree sorted-key DFS order, starting at index 3
  auto F = [&](int i) { return (const float*)d_in[i]; };
  auto bnp = [&](int base) {
    BNP r; r.b = F(base); r.g = F(base + 1); r.m = F(base + 2); r.v = F(base + 3);
    return r;
  };
  // bases: out1=3 out2=8 spconv1=13 spconv2=18 spconv3=23
  //        stem_res1=28 stem_res2=40 subm1_bn=52 subm1_w=56
  //        unet1=57 unet2=96 unet3=135

  const Dim R0{49, 128, 128, 49 * 128 * 128};
  const Dim R1{25, 64, 64, 25 * 64 * 64};
  const Dim R2{13, 32, 32, 13 * 32 * 32};
  const Dim R3{7, 16, 16, 7 * 16 * 16};
  const Dim R4{4, 8, 8, 4 * 8 * 8};
  const Dim RO1{3, 16, 16, 3 * 16 * 16};
  const Dim RO2{1, 16, 16, 1 * 16 * 16};

  unsigned char* wsp = (unsigned char*)d_ws;
  size_t cur = 0;
  auto alloc = [&](size_t bytes) -> void* {
    void* r = wsp + cur;
    cur += (bytes + 255) & ~(size_t)255;
    return r;
  };
  int* winner = (int*)alloc((size_t)R0.V * 4);
  unsigned char* mask0 = (unsigned char*)alloc(R0.V);
  unsigned char* mask1 = (unsigned char*)alloc(R1.V);
  unsigned char* mask2 = (unsigned char*)alloc(R2.V);
  unsigned char* mask3 = (unsigned char*)alloc(R3.V);
  unsigned char* mask4 = (unsigned char*)alloc(R4.V);
  unsigned char* masko1 = (unsigned char*)alloc(RO1.V);
  unsigned char* masko2 = (unsigned char*)alloc(RO2.V);
  int* list0 = (int*)alloc((size_t)R0.V * 4);
  int* list1 = (int*)alloc((size_t)R1.V * 4);
  int* list2 = (int*)alloc((size_t)R2.V * 4);
  int* list3 = (int*)alloc((size_t)R3.V * 4);
  int* list4 = (int*)alloc((size_t)R4.V * 4);
  int* listo1 = (int*)alloc((size_t)RO1.V * 4);
  int* listo2 = (int*)alloc((size_t)RO2.V * 4);
  int* cnts = (int*)alloc(8 * 4);
  __bf16* wpack = (__bf16*)alloc(64 * 27 * 64 * 2 + 4096);  // max packed B

  float* in0 = (float*)alloc((size_t)R0.V * 16 * 4);  // zero-padded 4->16 ch
  float* S0 = (float*)alloc((size_t)R0.V * 16 * 4);
  float* S1 = (float*)alloc((size_t)R0.V * 16 * 4);
  float* S2 = (float*)alloc((size_t)R0.V * 16 * 4);
  float* U1a = (float*)alloc((size_t)R1.V * 32 * 4);
  float* U1b = (float*)alloc((size_t)R1.V * 32 * 4);
  float* U1c = (float*)alloc((size_t)R1.V * 32 * 4);
  float* U1d = (float*)alloc((size_t)R1.V * 32 * 4);
  float* D2a = (float*)alloc((size_t)R2.V * 32 * 4);
  float* D2b = (float*)alloc((size_t)R2.V * 32 * 4);
  float* D2c = (float*)alloc((size_t)R2.V * 32 * 4);
  float* E2a = (float*)alloc((size_t)R2.V * 64 * 4);
  float* E2b = (float*)alloc((size_t)R2.V * 64 * 4);
  float* E2c = (float*)alloc((size_t)R2.V * 64 * 4);
  float* E2d = (float*)alloc((size_t)R2.V * 64 * 4);
  float* F3a = (float*)alloc((size_t)R3.V * 64 * 4);
  float* F3b = (float*)alloc((size_t)R3.V * 64 * 4);
  float* F3c = (float*)alloc((size_t)R3.V * 64 * 4);
  float* G3a = (float*)alloc((size_t)R3.V * 64 * 4);
  float* G3b = (float*)alloc((size_t)R3.V * 64 * 4);
  float* G3c = (float*)alloc((size_t)R3.V * 64 * 4);
  float* G3d = (float*)alloc((size_t)R3.V * 64 * 4);
  float* H4a = (float*)alloc((size_t)R4.V * 64 * 4);
  float* H4b = (float*)alloc((size_t)R4.V * 64 * 4);
  float* H4c = (float*)alloc((size_t)R4.V * 64 * 4);
  float* O1 = (float*)alloc((size_t)RO1.V * 64 * 4);
  float* O2 = (float*)alloc((size_t)RO2.V * 64 * 4);

  // ---- launch helpers ----
  // lc = log2(padded Cin); k311: kernel (3,1,1) instead of (3,3,3)
  auto conv = [&](const float* src, Dim di, int CinReal, int lc, const float* w,
                  const float* bias, float* dst, Dim dd, int Cout,
                  const int* list, const int* cnt, int maxCnt, bool k311,
                  int sz, int sy, int sx, int pz, int py, int px, int dil) {
    int taps = k311 ? 3 : 27;
    int ksteps = ((taps << lc) + 31) / 32;
    int ptotal = ksteps * (Cout >> 4) * 32 * 16;
    pack_weights_kernel<<<(ptotal + 255) / 256, 256, 0, stream>>>(
        w, CinReal, lc, taps, Cout, ksteps, wpack);
    hipMemsetAsync(dst, 0, (size_t)dd.V * Cout * 4, stream);
    int blocks = ((maxCnt + 15) / 16 + 3) / 4;  // 4 waves / block
#define CARGS src, di.D, di.H, di.W, wpack, bias, dst, dd.H, dd.W, list, cnt, \
              dil, sz, sy, sx, pz, py, px
    if (k311)                        conv_wmma<6, 3, 1, 1, 4><<<blocks, 128, 0, stream>>>(CARGS);
    else if (lc == 4 && Cout == 16)  conv_wmma<4, 3, 3, 3, 1><<<blocks, 128, 0, stream>>>(CARGS);
    else if (lc == 4 && Cout == 32)  conv_wmma<4, 3, 3, 3, 2><<<blocks, 128, 0, stream>>>(CARGS);
    else if (lc == 5 && Cout == 32)  conv_wmma<5, 3, 3, 3, 2><<<blocks, 128, 0, stream>>>(CARGS);
    else if (lc == 5 && Cout == 64)  conv_wmma<5, 3, 3, 3, 4><<<blocks, 128, 0, stream>>>(CARGS);
    else                             conv_wmma<6, 3, 3, 3, 4><<<blocks, 128, 0, stream>>>(CARGS);
#undef CARGS
  };
  auto epi = [&](float* x, const float* before, const float* after, BNP bn,
                 const int* list, const int* cnt, int maxCnt, int lcC, int relu) {
    int blocks = ((maxCnt << lcC) + 255) / 256;
    bn_epilogue<<<blocks, 256, 0, stream>>>(x, before, after, bn.g, bn.b, bn.m,
                                            bn.v, list, cnt, lcC, relu);
  };
  auto pool = [&](const unsigned char* sm, Dim di, unsigned char* dm, Dim dd,
                  int* list, int* cnt, int kz, int ky, int kx, int sz, int sy,
                  int sx, int pz, int py, int px) {
    int blocks = (dd.V + 255) / 256;
    maskpool_kernel<<<blocks, 256, 0, stream>>>(sm, di.D, di.H, di.W, dm, dd.D,
                                                dd.H, dd.W, list, cnt, kz, ky,
                                                kx, sz, sy, sx, pz, py, px);
  };
  auto resblock = [&](float* X, float* T1, float* T2, Dim d, int lcC,
                      const int* list, const int* cnt, int maxC, int rb) {
    int C = 1 << lcC;
    conv(X, d, C, lcC, F(rb + 10), F(rb + 0), T1, d, C, list, cnt, maxC, false,
         1, 1, 1, 1, 1, 1, 0);
    epi(T1, nullptr, nullptr, bnp(rb + 2), list, cnt, maxC, lcC, 1);
    conv(T1, d, C, lcC, F(rb + 11), F(rb + 1), T2, d, C, list, cnt, maxC, false,
         1, 1, 1, 1, 1, 1, 0);
    epi(T2, nullptr, X, bnp(rb + 6), list, cnt, maxC, lcC, 1);
    return T2;
  };
  auto unet = [&](float* X, Dim r, Dim r2, int lcC,
                  const int* lR, const int* cR, int mR,
                  const int* lR2, const int* cR2, int mR2,
                  float* T1, float* T2, float* Y, float* Za, float* Zb,
                  float* T3, int base) -> float* {
    int C = 1 << lcC;
    float* f0 = resblock(X, T1, T2, r, lcC, lR, cR, mR, base + 9);
    conv(f0, r, C, lcC, F(base + 26), F(base + 21), Y, r2, C, lR2, cR2, mR2,
         false, 2, 2, 2, 1, 1, 1, 0);                        // enc1_down (bias)
    epi(Y, nullptr, nullptr, bnp(base + 22), lR2, cR2, mR2, lcC, 1);
    float* y = resblock(Y, Za, Zb, r2, lcC, lR2, cR2, mR2, base + 27);
    conv(y, r2, C, lcC, F(base + 8), nullptr, T3, r, C, lR, cR, mR, false,
         1, 1, 1, 1, 1, 1, 1);                               // inverse conv
    epi(T3, nullptr, nullptr, bnp(base), lR, cR, mR, lcC, 1);
    epi(T3, f0, nullptr, bnp(base + 4), lR, cR, mR, lcC, 0);  // decoder_norm
    return T3;
  };

  // ---- dense input + masks + active lists ----
  hipMemsetAsync(winner, 0xFF, (size_t)R0.V * 4, stream);
  scatter_max_kernel<<<(nvox + 255) / 256, 256, 0, stream>>>(coords, nvox, 128,
                                                             128, winner);
  hipMemsetAsync(in0, 0, (size_t)R0.V * 16 * 4, stream);
  scatter_write_kernel<<<(nvox + 255) / 256, 256, 0, stream>>>(
      vf, coords, nvox, 128, 128, winner, in0);
  hipMemsetAsync(cnts, 0, 8 * 4, stream);
  compact_kernel<<<(R0.V + 255) / 256, 256, 0, stream>>>(winner, R0.V, mask0,
                                                         list0, cnts + 0);
  pool(mask0, R0, mask1, R1, list1, cnts + 1, 3, 3, 3, 2, 2, 2, 1, 1, 1);
  pool(mask1, R1, mask2, R2, list2, cnts + 2, 3, 3, 3, 2, 2, 2, 1, 1, 1);
  pool(mask2, R2, mask3, R3, list3, cnts + 3, 3, 3, 3, 2, 2, 2, 1, 1, 1);
  pool(mask3, R3, mask4, R4, list4, cnts + 4, 3, 3, 3, 2, 2, 2, 1, 1, 1);
  pool(mask3, R3, masko1, RO1, listo1, cnts + 5, 3, 1, 1, 2, 1, 1, 0, 0, 0);
  pool(masko1, RO1, masko2, RO2, listo2, cnts + 6, 3, 1, 1, 2, 1, 1, 0, 0, 0);

  // ---- stem ----
  conv(in0, R0, 4, 4, F(56), nullptr, S0, R0, 16, list0, cnts + 0, nvox, false,
       1, 1, 1, 1, 1, 1, 0);                                 // subm1 (4->16)
  epi(S0, nullptr, nullptr, bnp(52), list0, cnts + 0, nvox, 4, 1);
  resblock(S0, S1, S2, R0, 4, list0, cnts + 0, nvox, 28);    // stem_res1
  resblock(S2, S0, S1, R0, 4, list0, cnts + 0, nvox, 40);    // stem_res2
  conv(S1, R0, 16, 4, F(17), nullptr, U1a, R1, 32, list1, cnts + 1, R1.V, false,
       2, 2, 2, 1, 1, 1, 0);                                 // spconv1
  epi(U1a, nullptr, nullptr, bnp(13), list1, cnts + 1, R1.V, 5, 1);

  // ---- conv2: UNet(32) + spconv2 ----
  float* u1 = unet(U1a, R1, R2, 5, list1, cnts + 1, R1.V, list2, cnts + 2, R2.V,
                   U1b, U1c, D2a, D2b, D2c, U1d, 57);
  conv(u1, R1, 32, 5, F(22), nullptr, E2a, R2, 64, list2, cnts + 2, R2.V, false,
       2, 2, 2, 1, 1, 1, 0);                                 // spconv2
  epi(E2a, nullptr, nullptr, bnp(18), list2, cnts + 2, R2.V, 6, 1);

  // ---- conv3: UNet(64) + spconv3 ----
  float* u2 = unet(E2a, R2, R3, 6, list2, cnts + 2, R2.V, list3, cnts + 3, R3.V,
                   E2b, E2c, F3a, F3b, F3c, E2d, 96);
  conv(u2, R2, 64, 6, F(27), nullptr, G3a, R3, 64, list3, cnts + 3, R3.V, false,
       2, 2, 2, 1, 1, 1, 0);                                 // spconv3
  epi(G3a, nullptr, nullptr, bnp(23), list3, cnts + 3, R3.V, 6, 1);

  // ---- layers: UNet(64) ----
  float* u3 = unet(G3a, R3, R4, 6, list3, cnts + 3, R3.V, list4, cnts + 4, R4.V,
                   G3b, G3c, H4a, H4b, H4c, G3d, 135);

  // ---- conv_out ----
  conv(u3, R3, 64, 6, F(7), nullptr, O1, RO1, 64, listo1, cnts + 5, RO1.V, true,
       2, 1, 1, 0, 0, 0, 0);                                 // out1 (3,1,1)
  epi(O1, nullptr, nullptr, bnp(3), listo1, cnts + 5, RO1.V, 6, 1);
  conv(O1, RO1, 64, 6, F(12), nullptr, O2, RO2, 64, listo2, cnts + 6, RO2.V, true,
       2, 1, 1, 0, 0, 0, 0);                                 // out2
  epi(O2, nullptr, nullptr, bnp(8), listo2, cnts + 6, RO2.V, 6, 1);

  write_out_kernel<<<(64 * 256 + 255) / 256, 256, 0, stream>>>(O2,
                                                               (float*)d_out);
}
